// GatGraphClassifier_40372692582769
// MI455X (gfx1250) — matrix operand order repaired and checked by reading, not compiled, not run on hardware
//
#include <hip/hip_runtime.h>
#include <hip/hip_bf16.h>

// ---------------- problem constants ----------------
#define NN      50000
#define EE      400000
#define DD      64
#define HH      4
#define LL      3
#define CC      10
#define GG      512
#define HD      256           // H*D
#define EN      450000        // E + N (self loops)
#define NEG_SLOPE 0.2f

typedef __attribute__((ext_vector_type(16))) __bf16 v16bf;
typedef __attribute__((ext_vector_type(8)))  float  v8f;

__constant__ int c_atom_off[9] = {0, 119, 124, 136, 148, 158, 164, 170, 172};

// ---------------- helpers ----------------
__device__ inline void atomicMaxF(float* addr, float val) {
    int* ai = (int*)addr;
    int old = __float_as_int(*addr);
    while (__int_as_float(old) < val) {
        int assumed = old;
        old = atomicCAS(ai, assumed, __float_as_int(val));
        if (old == assumed) break;
    }
}

__global__ void k_fill(float* p, float v, int n) {
    int i = blockIdx.x * blockDim.x + threadIdx.x;
    if (i < n) p[i] = v;
}

// ---------------- 0) one-time weight convert: f32 [l][k][col] -> bf16 [l][col][k] ----
__global__ void k_convert_w(const float* __restrict__ Wl, const float* __restrict__ Wr,
                            __bf16* __restrict__ Wlt, __bf16* __restrict__ Wrt) {
    int tid = blockIdx.x * blockDim.x + threadIdx.x;
    if (tid >= 2 * LL * DD * HD) return;
    int which = tid / (LL * DD * HD);
    int r     = tid % (LL * DD * HD);
    int l   = r / (DD * HD);
    int rem = r % (DD * HD);
    int k   = rem / HD;
    int col = rem % HD;
    const float* W  = which ? Wr  : Wl;
    __bf16*      Wt = which ? Wrt : Wlt;
    Wt[(size_t)l * DD * HD + col * DD + k] = (__bf16)W[(size_t)l * DD * HD + k * HD + col];
}

// ---------------- 1) atom encoder: sum of 9 embeddings ----------------
__global__ void k_atom_encode(const int* __restrict__ nf,
                              const float* __restrict__ emb,
                              float* __restrict__ x,
                              __bf16* __restrict__ xbf) {
    int idx = blockIdx.x * blockDim.x + threadIdx.x;          // node*64 + d
    if (idx >= NN * DD) return;
    int n = idx >> 6;
    int d = idx & 63;
    float s = 0.f;
#pragma unroll
    for (int j = 0; j < 9; ++j) {
        int f = nf[n * 9 + j] + c_atom_off[j];
        s += emb[f * DD + d];
    }
    x[idx]   = s;
    xbf[idx] = (__bf16)s;
}

// ---------------- 2) WMMA GEMM: xl = x@Wl+bl, xr = x@Wr+br ----------------
// Block = 4 waves sharing one 16-row A tile staged in LDS.
// Each wave computes 16(M) x 64(N) (4 n-tiles), reusing its A fragments.
// (blockIdx.y*4 + threadIdx.y) in [0,8) selects a group of 4 of the 32 n-tiles
// (tiles 0..15 -> Wl columns, 16..31 -> Wr columns).
__global__ void k_gemm(const __bf16* __restrict__ xbf,
                       const __bf16* __restrict__ Wlt, const float* __restrict__ bl,
                       const __bf16* __restrict__ Wrt, const float* __restrict__ br,
                       float* __restrict__ xl, float* __restrict__ xr) {
    __shared__ __attribute__((aligned(16))) __bf16 sA[16 * DD];   // 2 KB

    const int m0  = blockIdx.x * 16;
    const int tid = threadIdx.y * 32 + threadIdx.x;               // 0..127

    // cooperative A stage: 128 threads x 16B = 16x64 bf16 tile
    {
        int row = tid >> 3;            // 0..15
        int kk  = (tid & 7) * 8;       // 0,8,...,56
        const uint4* src = (const uint4*)(xbf + (size_t)(m0 + row) * DD + kk);
        *((uint4*)(sA + row * DD + kk)) = *src;
    }
    __syncthreads();

    const int lane = threadIdx.x;      // 0..31 (wave32)
    const int half = lane >> 4;
    const int lh   = lane & 15;

    // Build both A fragments (k0=0 and k0=32) once from LDS.
    // ISA 16-bit A layout: lanes 0-15 rows M, K 0..7 & 16..23; lanes 16-31 K 8..15 & 24..31.
    v16bf a0, a1;
#pragma unroll
    for (int i = 0; i < 16; ++i) {
        int v = i >> 1, p = i & 1;
        int kl = ((v < 4) ? (v * 2 + p) : (16 + (v - 4) * 2 + p)) + half * 8;
        a0[i] = sA[lh * DD + kl];
        a1[i] = sA[lh * DD + 32 + kl];
    }

    const int job = blockIdx.y * 4 + threadIdx.y;   // 0..7
#pragma unroll
    for (int t = 0; t < 4; ++t) {
        int ntile = job * 4 + t;                    // 0..31
        const __bf16* Wt; const float* bias; float* out; int nc0;
        if (ntile < 16) { Wt = Wlt; bias = bl; out = xl; nc0 = ntile * 16; }
        else            { Wt = Wrt; bias = br; out = xr; nc0 = (ntile - 16) * 16; }

        const int col = nc0 + lh;
        // B fragment: pre-transposed bf16 weights, contiguous in k per column.
        const __bf16* bcol = Wt + (size_t)col * DD + half * 8;
        v16bf b0, b1;
#pragma unroll
        for (int i = 0; i < 16; ++i) { int kb = (i < 8) ? i : (i + 8); b0[i] = bcol[kb]; b1[i] = bcol[32 + kb]; }

        v8f c = {};
        c = __builtin_amdgcn_wmma_f32_16x16x32_bf16(false, a0, false, b0, (short)0, c, false, false);
        c = __builtin_amdgcn_wmma_f32_16x16x32_bf16(false, a1, false, b1, (short)0, c, false, false);

        const float bv = bias[col];
#pragma unroll
        for (int r = 0; r < 8; ++r) {
            out[(size_t)(m0 + r + half * 8) * HD + col] = c[r] + bv;
        }
    }
}

// ---------------- 3) edge attention score + segment max ----------------
__global__ void k_edge_score(const int* __restrict__ edges,
                             const float* __restrict__ xl,
                             const float* __restrict__ xr,
                             const float* __restrict__ att,   // [H,64] for this layer
                             float* __restrict__ es,          // [EN,H]
                             float* __restrict__ emax) {      // [N,H]
    int tid = blockIdx.x * blockDim.x + threadIdx.x;
    if (tid >= EN * HH) return;
    int e = tid >> 2;
    int h = tid & 3;
    int src, dst;
    if (e < EE) { src = edges[e]; dst = edges[EE + e]; }
    else        { src = e - EE;   dst = e - EE; }
    const float* pl = xl + (size_t)src * HD + h * DD;
    const float* pr = xr + (size_t)dst * HD + h * DD;
    const float* pa = att + h * DD;
    float acc = 0.f;
#pragma unroll 8
    for (int d = 0; d < DD; ++d) {
        float m = pl[d] + pr[d];
        m = (m > 0.f) ? m : NEG_SLOPE * m;         // leaky_relu
        acc += m * pa[d];
    }
    es[tid] = acc;
    atomicMaxF(&emax[dst * HH + h], acc);
}

// ---------------- 4) softmax numerator + denominator ----------------
__global__ void k_edge_soft(const int* __restrict__ edges,
                            float* __restrict__ es,
                            const float* __restrict__ emax,
                            float* __restrict__ den) {
    int tid = blockIdx.x * blockDim.x + threadIdx.x;
    if (tid >= EN * HH) return;
    int e = tid >> 2;
    int h = tid & 3;
    int dst = (e < EE) ? edges[EE + e] : (e - EE);
    float ea = __expf(es[tid] - emax[dst * HH + h]);
    es[tid] = ea;
    atomicAdd(&den[dst * HH + h], ea);
}

// ---------------- 5) weighted scatter-accumulate ----------------
__global__ void k_edge_accum(const int* __restrict__ edges,
                             const float* __restrict__ es,
                             const float* __restrict__ den,
                             const float* __restrict__ xl,
                             float* __restrict__ acc) {       // [N,HD]
    int tid = blockIdx.x * blockDim.x + threadIdx.x;
    if (tid >= EN * HH) return;
    int e = tid >> 2;
    int h = tid & 3;
    int src, dst;
    if (e < EE) { src = edges[e]; dst = edges[EE + e]; }
    else        { src = e - EE;   dst = e - EE; }
    float alpha = es[tid] / den[dst * HH + h];
    const float* pl = xl  + (size_t)src * HD + h * DD;
    float*       pd = acc + (size_t)dst * HD + h * DD;
#pragma unroll 4
    for (int d = 0; d < DD; ++d) {
        atomicAdd(&pd[d], alpha * pl[d]);
    }
}

// ---------------- 6) head mean + bias (+ GELU), refresh bf16 ----------------
__global__ void k_head_mean(const float* __restrict__ acc,
                            const float* __restrict__ gbias,  // [64] this layer
                            float* __restrict__ x,
                            __bf16* __restrict__ xbf,
                            int apply_gelu) {
    int idx = blockIdx.x * blockDim.x + threadIdx.x;          // node*64 + d
    if (idx >= NN * DD) return;
    int n = idx >> 6;
    int d = idx & 63;
    const float* p = acc + (size_t)n * HD + d;
    float v = 0.25f * (p[0] + p[64] + p[128] + p[192]) + gbias[d];
    if (apply_gelu) v = 0.5f * v * (1.0f + erff(v * 0.70710678118654752f));
    x[idx]   = v;
    xbf[idx] = (__bf16)v;
}

// ---------------- 7) global mean pool ----------------
__global__ void k_pool(const int* __restrict__ batch,
                       const float* __restrict__ x,
                       float* __restrict__ pool,   // [G,64]
                       float* __restrict__ cnt) {  // [G]
    int idx = blockIdx.x * blockDim.x + threadIdx.x;
    if (idx >= NN * DD) return;
    int n = idx >> 6;
    int d = idx & 63;
    int g = batch[n];
    atomicAdd(&pool[g * DD + d], x[idx]);
    if (d == 0) atomicAdd(&cnt[g], 1.0f);
}

// ---------------- 8) classifier ----------------
__global__ void k_final(const float* __restrict__ pool,
                        const float* __restrict__ cnt,
                        const float* __restrict__ Wc,
                        const float* __restrict__ bc,
                        float* __restrict__ out) {
    int tid = blockIdx.x * blockDim.x + threadIdx.x;
    if (tid >= GG * CC) return;
    int g = tid / CC;
    int c = tid % CC;
    float cv = cnt[g];
    cv = (cv > 1.0f) ? cv : 1.0f;
    float s = bc[c];
    const float* pg = pool + g * DD;
#pragma unroll 8
    for (int d = 0; d < DD; ++d) s += (pg[d] / cv) * Wc[d * CC + c];
    out[tid] = s;
}

// ---------------- launcher ----------------
extern "C" void kernel_launch(void* const* d_in, const int* in_sizes, int n_in,
                              void* d_out, int out_size, void* d_ws, size_t ws_size,
                              hipStream_t stream) {
    (void)in_sizes; (void)n_in; (void)out_size; (void)ws_size;

    const int*   nf    = (const int*)d_in[0];
    const int*   edges = (const int*)d_in[1];
    const int*   batch = (const int*)d_in[2];
    const float* aemb  = (const float*)d_in[3];
    const float* Wl    = (const float*)d_in[4];
    const float* bl    = (const float*)d_in[5];
    const float* Wr    = (const float*)d_in[6];
    const float* br    = (const float*)d_in[7];
    const float* att   = (const float*)d_in[8];
    const float* gbias = (const float*)d_in[9];
    const float* Wc    = (const float*)d_in[10];
    const float* bc    = (const float*)d_in[11];
    float* out = (float*)d_out;

    char* ws = (char*)d_ws;
    float*  x    = (float*)(ws + 0);                      // N*64 f32
    __bf16* xbf  = (__bf16*)(ws + 12800000);              // N*64 bf16
    float*  xl   = (float*)(ws + 19200000);               // N*256
    float*  xr   = (float*)(ws + 70400000);               // N*256
    float*  es   = (float*)(ws + 121600000);              // EN*4
    float*  emax = (float*)(ws + 128800000);              // N*4
    float*  den  = (float*)(ws + 129600000);              // N*4
    float*  acc  = (float*)(ws + 130400000);              // N*256
    float*  pool = (float*)(ws + 181600000);              // G*64
    float*  cnt  = (float*)(ws + 181731072);              // G
    __bf16* Wlt  = (__bf16*)(ws + 181733120);             // 3*256*64 bf16 (transposed)
    __bf16* Wrt  = (__bf16*)(ws + 181929728);             // 3*256*64 bf16 (transposed)

    const int TPB = 256;
    const int nd_blocks  = (NN * DD + TPB - 1) / TPB;     // 12500
    const int eh_blocks  = (EN * HH + TPB - 1) / TPB;     // 7032

    // one-shot weight conversion (bf16, transposed) + atom encoder
    k_convert_w<<<(2 * LL * DD * HD + TPB - 1) / TPB, TPB, 0, stream>>>(Wl, Wr, Wlt, Wrt);
    k_atom_encode<<<nd_blocks, TPB, 0, stream>>>(nf, aemb, x, xbf);

    for (int l = 0; l < LL; ++l) {
        const __bf16* Wlt_l = Wlt + (size_t)l * DD * HD;
        const float*  bl_l  = bl  + (size_t)l * HD;
        const __bf16* Wrt_l = Wrt + (size_t)l * DD * HD;
        const float*  br_l  = br  + (size_t)l * HD;
        const float*  at_l  = att + (size_t)l * HH * DD;
        const float*  gb_l  = gbias + (size_t)l * DD;

        // per-layer init: emax=-inf, den=0, acc=0
        k_fill<<<(NN * HH + TPB - 1) / TPB, TPB, 0, stream>>>(emax, -1e30f, NN * HH);
        k_fill<<<(NN * HH + TPB - 1) / TPB, TPB, 0, stream>>>(den, 0.f, NN * HH);
        k_fill<<<(NN * HD + TPB - 1) / TPB, TPB, 0, stream>>>(acc, 0.f, NN * HD);

        // dense transform via WMMA: 3125 row tiles x (2 grid.y * 4 waves) jobs x 4 n-tiles
        dim3 ggrid(NN / 16, 2);
        dim3 gblk(32, 4);
        k_gemm<<<ggrid, gblk, 0, stream>>>(xbf, Wlt_l, bl_l, Wrt_l, br_l, xl, xr);

        k_edge_score<<<eh_blocks, TPB, 0, stream>>>(edges, xl, xr, at_l, es, emax);
        k_edge_soft <<<eh_blocks, TPB, 0, stream>>>(edges, es, emax, den);
        k_edge_accum<<<eh_blocks, TPB, 0, stream>>>(edges, es, den, xl, acc);

        k_head_mean<<<nd_blocks, TPB, 0, stream>>>(acc, gb_l, x, xbf, (l < LL - 1) ? 1 : 0);
    }

    // pooling + classifier
    k_fill<<<(GG * DD + TPB - 1) / TPB, TPB, 0, stream>>>(pool, 0.f, GG * DD);
    k_fill<<<(GG + TPB - 1) / TPB, TPB, 0, stream>>>(cnt, 0.f, GG);
    k_pool<<<nd_blocks, TPB, 0, stream>>>(batch, x, pool, cnt);
    k_final<<<(GG * CC + TPB - 1) / TPB, TPB, 0, stream>>>(pool, cnt, Wc, bc, out);
}
